// IttAttention_30640296690473
// MI455X (gfx1250) — compile-verified, matrix operand order checked by reading, not run on hardware
//
#include <hip/hip_runtime.h>
#include <hip/hip_bf16.h>

// Problem constants
#define Bc 2
#define Sc 2048
#define Dc 1024
#define Hc 16
#define DHc 64

typedef unsigned short u16t;
typedef __attribute__((ext_vector_type(16))) __bf16 v16bf;
typedef __attribute__((ext_vector_type(8)))  float  v8f;
typedef __attribute__((ext_vector_type(4)))  unsigned int v4u;

union BF16Tile {
  v16bf v;
  v4u   q[2];
  u16t  h[16];
};

__device__ __forceinline__ u16t f32_to_bf16(float f) {
  unsigned int u = __float_as_uint(f);
  u += 0x7FFFu + ((u >> 16) & 1u);   // round-to-nearest-even
  return (u16t)(u >> 16);
}

__device__ __forceinline__ v8f wmma_bf16(const BF16Tile& a, const BF16Tile& b, v8f c) {
  // D(16x16,f32) = A(16x32,bf16) * B(32x16,bf16) + C
  return __builtin_amdgcn_wmma_f32_16x16x32_bf16(false, a.v, false, b.v,
                                                 (short)0, c, false, false);
}

// ---------------------------------------------------------------------------
// K0: convert hidden_states -> bf16, and W{q,k,v,o} -> transposed bf16 [N,K]
// ---------------------------------------------------------------------------
__global__ void k0_convert(const float* __restrict__ hs,
                           const float* __restrict__ Wq, const float* __restrict__ Wk,
                           const float* __restrict__ Wv, const float* __restrict__ Wo,
                           u16t* __restrict__ hsb,
                           u16t* __restrict__ Wqt, u16t* __restrict__ Wkt,
                           u16t* __restrict__ Wvt, u16t* __restrict__ Wot) {
  const int N0 = Bc * Sc * Dc;                    // 4194304
  int i = blockIdx.x * blockDim.x + threadIdx.x;
  if (i < N0) {
    hsb[i] = f32_to_bf16(hs[i]);
  } else {
    int j = i - N0;
    if (j >= 4 * Dc * Dc) return;
    int mat = j >> 20;                            // Dc*Dc = 2^20
    int idx = j & 0xFFFFF;
    int n = idx >> 10, k = idx & 1023;
    const float* W = (mat == 0) ? Wq : (mat == 1) ? Wk : (mat == 2) ? Wv : Wo;
    u16t*        T = (mat == 0) ? Wqt : (mat == 1) ? Wkt : (mat == 2) ? Wvt : Wot;
    T[(size_t)n * Dc + k] = f32_to_bf16(W[(size_t)k * Dc + n]);
  }
}

// ---------------------------------------------------------------------------
// K1: fused QKV projection. C[16,16] tiles via bf16 WMMA, f32 accumulate.
//   proj 0 -> Q  stored [B,H,S,DH] bf16
//   proj 1 -> K  stored [B,H,S,DH] bf16
//   proj 2 -> V  stored [B,H,DH,S] bf16 (transposed for the P*V B-operand)
// ---------------------------------------------------------------------------
__global__ void k1_qkv(const u16t* __restrict__ hsb,
                       const u16t* __restrict__ Wqt, const u16t* __restrict__ Wkt,
                       const u16t* __restrict__ Wvt,
                       const float* __restrict__ bq, const float* __restrict__ bk,
                       const float* __restrict__ bv,
                       u16t* __restrict__ qb, u16t* __restrict__ kb,
                       u16t* __restrict__ vt) {
  const int lane = threadIdx.x & 31;
  const int wave = threadIdx.x >> 5;
  const int m0   = blockIdx.x * 16;
  const int proj = blockIdx.y;
  const int n0   = (blockIdx.z * 8 + wave) * 16;

  const u16t*  Wt   = (proj == 0) ? Wqt : (proj == 1) ? Wkt : Wvt;
  const float* bias = (proj == 0) ? bq  : (proj == 1) ? bk  : bv;

  const int l15  = lane & 15;
  const int khi  = (lane < 16) ? 0 : 8;   // K sub-block per ISA A/B layout
  const int rbase= (lane < 16) ? 0 : 8;   // C/D row base per ISA layout
  const int arow = m0 + l15;
  const int bcol = n0 + l15;

  v8f acc = {};
  for (int kc = 0; kc < Dc; kc += 32) {
    BF16Tile A, Bt;
    const u16t* ap = hsb + (size_t)arow * Dc + kc + khi;
    A.q[0] = *(const v4u*)ap;
    A.q[1] = *(const v4u*)(ap + 16);
    const u16t* bp = Wt + (size_t)bcol * Dc + kc + khi;
    Bt.q[0] = *(const v4u*)bp;
    Bt.q[1] = *(const v4u*)(bp + 16);
    acc = wmma_bf16(A, Bt, acc);
  }

  const float bval = bias[bcol];
  const int h = bcol >> 6, dh = bcol & 63;
#pragma unroll
  for (int r = 0; r < 8; ++r) {
    const int row = m0 + rbase + r;
    const int b = row >> 11, s = row & 2047;
    const u16t x = f32_to_bf16(acc[r] + bval);
    if (proj == 2)
      vt[((size_t)(b * Hc + h) * DHc + dh) * Sc + s] = x;
    else if (proj == 0)
      qb[((size_t)(b * Hc + h) * Sc + s) * DHc + dh] = x;
    else
      kb[((size_t)(b * Hc + h) * Sc + s) * DHc + dh] = x;
  }
}

// ---------------------------------------------------------------------------
// K2: attention for one (b,h,16-query tile).
//   Phase 1: scores[16,2048] = Q*K^T/8 + mask  (WMMA) -> LDS (128 KB)
//   Phase 2: softmax in LDS; single contiguous probs write to HBM
//   Phase 3: ctx[16,64] = P*V (WMMA, P read from LDS) -> ctx bf16 [B,S,D]
// ---------------------------------------------------------------------------
__global__ void k2_attn(const u16t* __restrict__ qb, const u16t* __restrict__ kb,
                        const u16t* __restrict__ vt, const float* __restrict__ maskp,
                        float* __restrict__ probs, u16t* __restrict__ ctxb) {
  extern __shared__ char smem[];
  float* sc      = (float*)smem;          // 16*2048 f32 = 128 KB
  float* partial = sc + 16 * Sc;          // 2*4*256 f32 = 8 KB
  float* rowmax  = partial + 2048;        // 16
  float* rowinv  = rowmax + 16;           // 16

  const int tid  = threadIdx.x;
  const int lane = tid & 31;
  const int wave = tid >> 5;
  const int qt   = blockIdx.x & 127;
  const int bh   = blockIdx.x >> 7;       // 0..31
  const int b    = bh >> 4;
  const int q0   = qt * 16;

  const int l15   = lane & 15;
  const int khi   = (lane < 16) ? 0 : 8;
  const int rbase = (lane < 16) ? 0 : 8;

  // ---- Phase 1: scores ----
  const int arow = q0 + l15;
  for (int kt = wave; kt < Sc / 16; kt += 8) {
    const int k0 = kt * 16;
    v8f acc = {};
#pragma unroll
    for (int dc = 0; dc < DHc; dc += 32) {
      BF16Tile A, Bt;
      const u16t* qp = qb + ((size_t)bh * Sc + arow) * DHc + dc + khi;
      A.q[0] = *(const v4u*)qp;
      A.q[1] = *(const v4u*)(qp + 16);
      const u16t* kp = kb + ((size_t)bh * Sc + (k0 + l15)) * DHc + dc + khi;
      Bt.q[0] = *(const v4u*)kp;
      Bt.q[1] = *(const v4u*)(kp + 16);
      acc = wmma_bf16(A, Bt, acc);
    }
    const int   ncol = k0 + l15;
    const float m    = maskp[(size_t)b * Sc + ncol];
#pragma unroll
    for (int r = 0; r < 8; ++r)
      sc[(rbase + r) * Sc + ncol] = acc[r] * 0.125f + m;   // 1/sqrt(64)
  }
  __syncthreads();

  // ---- Phase 2a: per-row max & sum (16 threads per row, shuffle reduce) ----
  {
    const int g = tid >> 4, sg = tid & 15;
    float mx = -3.402823466e38f;
    for (int c = sg; c < Sc; c += 16) mx = fmaxf(mx, sc[g * Sc + c]);
#pragma unroll
    for (int off = 1; off < 16; off <<= 1) mx = fmaxf(mx, __shfl_xor(mx, off, 32));
    float sum = 0.f;
    for (int c = sg; c < Sc; c += 16) sum += __expf(sc[g * Sc + c] - mx);
#pragma unroll
    for (int off = 1; off < 16; off <<= 1) sum += __shfl_xor(sum, off, 32);
    if (sg == 0) { rowmax[g] = mx; rowinv[g] = 1.0f / sum; }
  }
  __syncthreads();

  // ---- Phase 2b: normalize; write probs once (contiguous); keep P in LDS ----
  {
    const size_t probBase = ((size_t)bh * Sc + q0) * Sc;
    for (int i = tid; i < 16 * Sc; i += 256) {
      const int r = i >> 11, c = i & (Sc - 1);
      const float p = __expf(sc[i] - rowmax[r]) * rowinv[r];
      sc[i] = p;
      probs[probBase + (size_t)r * Sc + c] = p;
    }
  }
  __syncthreads();

  // ---- Phase 3: ctx = P @ V  (wave -> (dh-tile, key-half)) ----
  {
    const int nt    = wave & 3;
    const int kpart = wave >> 2;
    const int dhcol = nt * 16 + l15;
    v8f acc = {};
    for (int kc = kpart * (Sc / 2); kc < (kpart + 1) * (Sc / 2); kc += 32) {
      BF16Tile A, Bt;
      const float* pp = &sc[l15 * Sc + kc + khi];
#pragma unroll
      for (int j = 0; j < 8; ++j) A.h[j]     = f32_to_bf16(pp[j]);
#pragma unroll
      for (int j = 0; j < 8; ++j) A.h[8 + j] = f32_to_bf16(pp[16 + j]);
      const u16t* vp = vt + ((size_t)bh * DHc + dhcol) * Sc + kc + khi;
      Bt.q[0] = *(const v4u*)vp;
      Bt.q[1] = *(const v4u*)(vp + 16);
      acc = wmma_bf16(A, Bt, acc);
    }
    float* pbuf = partial + wave * 256;      // wave = kpart*4 + nt
#pragma unroll
    for (int r = 0; r < 8; ++r)
      pbuf[(rbase + r) * 16 + l15] = acc[r];
  }
  __syncthreads();

  if (wave < 4) {                            // combine halves, emit bf16 ctx
    const int h = bh & 15;
    for (int i = lane; i < 256; i += 32) {
      const float v2 = partial[wave * 256 + i] + partial[(wave + 4) * 256 + i];
      const int r = i >> 4, c = i & 15;
      ctxb[(size_t)(b * Sc + q0 + r) * Dc + h * DHc + wave * 16 + c] = f32_to_bf16(v2);
    }
  }
}

// ---------------------------------------------------------------------------
// K3: out = LN(ctx @ Wo + bo + hidden).  16 rows per block; full N in LDS.
// ---------------------------------------------------------------------------
__global__ void k3_out_ln(const u16t* __restrict__ ctxb, const u16t* __restrict__ Wot,
                          const float* __restrict__ bo, const float* __restrict__ hs,
                          const float* __restrict__ ln_w, const float* __restrict__ ln_b,
                          float* __restrict__ out) {
  extern __shared__ char smem[];
  float* htile = (float*)smem;       // 16*1024 f32 = 64 KB
  float* umean = htile + 16 * Dc;    // 16
  float* rstd  = umean + 16;         // 16

  const int tid  = threadIdx.x;
  const int lane = tid & 31;
  const int wave = tid >> 5;
  const int m0   = blockIdx.x * 16;
  const int l15   = lane & 15;
  const int khi   = (lane < 16) ? 0 : 8;
  const int rbase = (lane < 16) ? 0 : 8;

  for (int nt = wave; nt < Dc / 16; nt += 8) {
    const int n0 = nt * 16;
    v8f acc = {};
    for (int kc = 0; kc < Dc; kc += 32) {
      BF16Tile A, Bt;
      const u16t* ap = ctxb + (size_t)(m0 + l15) * Dc + kc + khi;
      A.q[0] = *(const v4u*)ap;
      A.q[1] = *(const v4u*)(ap + 16);
      const u16t* bp = Wot + (size_t)(n0 + l15) * Dc + kc + khi;
      Bt.q[0] = *(const v4u*)bp;
      Bt.q[1] = *(const v4u*)(bp + 16);
      acc = wmma_bf16(A, Bt, acc);
    }
    const int   col  = n0 + l15;
    const float bval = bo[col];
#pragma unroll
    for (int r = 0; r < 8; ++r) {
      const int row = m0 + rbase + r;
      htile[(rbase + r) * Dc + col] = acc[r] + bval + hs[(size_t)row * Dc + col];
    }
  }
  __syncthreads();

  {
    const int g = tid >> 4, sg = tid & 15;
    float s1 = 0.f, s2 = 0.f;
    for (int c = sg; c < Dc; c += 16) {
      const float x = htile[g * Dc + c];
      s1 += x; s2 += x * x;
    }
#pragma unroll
    for (int off = 1; off < 16; off <<= 1) {
      s1 += __shfl_xor(s1, off, 32);
      s2 += __shfl_xor(s2, off, 32);
    }
    if (sg == 0) {
      const float u   = s1 * (1.0f / Dc);
      const float var = s2 * (1.0f / Dc) - u * u;   // biased variance
      umean[g] = u;
      rstd[g]  = rsqrtf(var + 1e-12f);
    }
  }
  __syncthreads();

  for (int i = tid; i < 16 * Dc; i += 256) {
    const int r = i >> 10, c = i & (Dc - 1);
    out[(size_t)(m0 + r) * Dc + c] = ln_w[c] * ((htile[i] - umean[r]) * rstd[r]) + ln_b[c];
  }
}

// ---------------------------------------------------------------------------
extern "C" void kernel_launch(void* const* d_in, const int* in_sizes, int n_in,
                              void* d_out, int out_size, void* d_ws, size_t ws_size,
                              hipStream_t stream) {
  const float* hs   = (const float*)d_in[0];
  const float* mask = (const float*)d_in[1];
  const float* Wq   = (const float*)d_in[2];
  const float* bq   = (const float*)d_in[3];
  const float* Wk   = (const float*)d_in[4];
  const float* bk   = (const float*)d_in[5];
  const float* Wv   = (const float*)d_in[6];
  const float* bv   = (const float*)d_in[7];
  const float* Wo   = (const float*)d_in[8];
  const float* bo   = (const float*)d_in[9];
  const float* lnw  = (const float*)d_in[10];
  const float* lnb  = (const float*)d_in[11];

  float* out   = (float*)d_out;
  float* probs = out + (size_t)Bc * Sc * Dc;

  char* ws = (char*)d_ws;
  u16t* hsb  = (u16t*)(ws);
  u16t* Wqt  = (u16t*)(ws + ((size_t)8  << 20));
  u16t* Wkt  = (u16t*)(ws + ((size_t)10 << 20));
  u16t* Wvt  = (u16t*)(ws + ((size_t)12 << 20));
  u16t* Wot  = (u16t*)(ws + ((size_t)14 << 20));
  u16t* qb   = (u16t*)(ws + ((size_t)16 << 20));
  u16t* kb   = (u16t*)(ws + ((size_t)24 << 20));
  u16t* vt   = (u16t*)(ws + ((size_t)32 << 20));
  u16t* ctxb = (u16t*)(ws + ((size_t)40 << 20));   // ends at 48 MiB

  const size_t lds2 = (size_t)(16 * Sc + 2048 + 32) * sizeof(float);  // ~136 KB
  const size_t lds3 = (size_t)(16 * Dc + 32) * sizeof(float);         // ~64 KB
  hipFuncSetAttribute((const void*)k2_attn,
                      hipFuncAttributeMaxDynamicSharedMemorySize, (int)lds2);
  hipFuncSetAttribute((const void*)k3_out_ln,
                      hipFuncAttributeMaxDynamicSharedMemorySize, (int)lds3);

  k0_convert<<<32768, 256, 0, stream>>>(hs, Wq, Wk, Wv, Wo, hsb, Wqt, Wkt, Wvt, Wot);
  k1_qkv<<<dim3(256, 3, 8), 256, 0, stream>>>(hsb, Wqt, Wkt, Wvt, bq, bk, bv, qb, kb, vt);
  k2_attn<<<dim3(Bc * Hc * (Sc / 16)), 256, lds2, stream>>>(qb, kb, vt, mask, probs, ctxb);
  k3_out_ln<<<dim3((Bc * Sc) / 16), 256, lds3, stream>>>(ctxb, Wot, bo, hs, lnw, lnb, out);
}
